// ImageEncoderViTSecondHalf_57183194579236
// MI455X (gfx1250) — matrix operand order repaired and checked by reading, not compiled, not run on hardware
//
#include <hip/hip_runtime.h>
#include <math.h>

// ---------------------------------------------------------------------------
// Types for CDNA5 WMMA (gfx1250, wave32)
// ---------------------------------------------------------------------------
typedef __bf16 bf16;
typedef unsigned long long ull;
typedef __attribute__((ext_vector_type(16))) __bf16 v16bf;
typedef __attribute__((ext_vector_type(8)))  float  v8f;

#define TB 256
#define BM 128
#define BN 128
#define BK 32
#define LDT (BK + 8)   // LDS row stride (elements) for both A and BT tiles

// ---------------------------------------------------------------------------
// Fragment loaders per CDNA5 ISA VGPR layouts (CDNA5 ISA)
// A 16x32 bf16: lanes 0-15 row M=lane, K 0..7 & 16..23; lanes 16-31 same row,
//               K 8..15 & 24..31.          As layout: [BM][LDT] (k contiguous)
// B 32x16 bf16: lanes 0-15 col N=lane, K 0..7 & 16..23; lanes 16-31 col N=lane-16,
//               K 8..15 & 24..31.                  Bs layout: [BN][LDT] (k contiguous)
// -> both fragments are contiguous runs in LDS => ds_load_b128 pairs.
// C/D 16x16 f32: lane%16 = N, vgpr r -> M = r + (lane>=16 ? 8 : 0).
// ---------------------------------------------------------------------------
__device__ inline v16bf frag_a(const bf16* As, int mBase, int lane) {
  const bf16* p = As + (mBase + (lane & 15)) * LDT + ((lane >> 4) << 3);
  v16bf a;
#pragma unroll
  for (int e = 0; e < 8; ++e) a[e] = p[e];
#pragma unroll
  for (int e = 0; e < 8; ++e) a[8 + e] = p[16 + e];
  return a;
}

__device__ inline v16bf frag_b(const bf16* Bs, int nBase, int lane) {
  const bf16* p = Bs + (nBase + (lane & 15)) * LDT + ((lane >> 4) << 3);
  v16bf b;
#pragma unroll
  for (int e = 0; e < 8; ++e) b[e] = p[e];
#pragma unroll
  for (int e = 0; e < 8; ++e) b[8 + e] = p[16 + e];
  return b;
}

__device__ inline float gelu_exact(float x) {
  return 0.5f * x * (1.0f + erff(x * 0.70710678118654752f));
}

// ---------------------------------------------------------------------------
// Batched bf16 GEMM:  C[z] = alpha * A[z] @ B[z] (+ bias) (+ res) per MODE
//   A : [M,K] row-major bf16
//   BT: [N,K] row-major bf16  (i.e. B transposed; K contiguous)
// MODE 0: f32 out | 1: bf16 out | 2: bf16 out + GELU | 3: f32 out + residual
// Requires: lda, ldbt, K, N multiples of 4 (holds for all shapes here).
// Double-buffered software pipeline: global loads for tile kt+1 are issued
// before the WMMAs of tile kt; interior tiles take an unguarded path, edge
// tiles use clamped addresses + cndmask (no exec-mask divergence).
// ---------------------------------------------------------------------------
template <int MODE>
__global__ __launch_bounds__(TB)
void gemm_bf16(const bf16* __restrict__ A, int lda, long sA,
               const bf16* __restrict__ BT, int ldbt, long sB,
               void* __restrict__ Cv, int ldc, long sC,
               const float* __restrict__ bias,
               const float* __restrict__ res,
               int M, int N, int K, float alpha) {
  __shared__ bf16 As[2][BM * LDT];
  __shared__ bf16 Bs[2][BN * LDT];

  const int z = blockIdx.z;
  A += (long)z * sA;
  BT += (long)z * sB;
  const int m0 = blockIdx.y * BM;
  const int n0 = blockIdx.x * BN;
  const int tid = threadIdx.x;
  const int wave = tid >> 5, lane = tid & 31;
  const int wm = wave >> 2, wn = wave & 3;  // 2 x 4 wave grid

  // staging geometry: thread covers rows {rs, rs+64} x 8 k-elems at kc
  const int rs = tid >> 2;        // 0..63
  const int kc = (tid & 3) * 8;   // 0,8,16,24
  const bool mFull = (m0 + BM <= M);
  const bool nFull = (n0 + BN <= N);

  const bf16* Ap0 = A + (long)(m0 + rs) * lda + kc;
  const bf16* Ap1 = Ap0 + (long)64 * lda;
  const bf16* Bp0 = BT + (long)(n0 + rs) * ldbt + kc;
  const bf16* Bp1 = Bp0 + (long)64 * ldbt;

  v8f acc[4][2];
#pragma unroll
  for (int mt = 0; mt < 4; ++mt)
#pragma unroll
    for (int nt = 0; nt < 2; ++nt)
#pragma unroll
      for (int r = 0; r < 8; ++r) acc[mt][nt][r] = 0.0f;

  const int kTiles = (K + BK - 1) / BK;

  ull ra[2][2], rb[2][2];

  auto loadRegs = [&](int kt) {
    const int k0 = kt * BK;
    const bool kFull = (k0 + BK <= K);
    if (mFull && kFull) {
#pragma unroll
      for (int j = 0; j < 2; ++j) {
        ra[0][j] = *(const ull*)(Ap0 + k0 + 4 * j);
        ra[1][j] = *(const ull*)(Ap1 + k0 + 4 * j);
      }
    } else {
      const int gr0 = m0 + rs, gr1 = gr0 + 64;
      const int r0c = gr0 < M ? gr0 : M - 1;
      const int r1c = gr1 < M ? gr1 : M - 1;
#pragma unroll
      for (int j = 0; j < 2; ++j) {
        int kp = k0 + kc + 4 * j;
        int kpc = kp < K ? kp : K - 4;  // K % 4 == 0, clamp keeps load legal
        ull v0 = *(const ull*)(A + (long)r0c * lda + kpc);
        ull v1 = *(const ull*)(A + (long)r1c * lda + kpc);
        ra[0][j] = (gr0 < M && kp < K) ? v0 : 0ull;
        ra[1][j] = (gr1 < M && kp < K) ? v1 : 0ull;
      }
    }
    if (nFull && kFull) {
#pragma unroll
      for (int j = 0; j < 2; ++j) {
        rb[0][j] = *(const ull*)(Bp0 + k0 + 4 * j);
        rb[1][j] = *(const ull*)(Bp1 + k0 + 4 * j);
      }
    } else {
      const int gn0 = n0 + rs, gn1 = gn0 + 64;
      const int n0c = gn0 < N ? gn0 : N - 1;
      const int n1c = gn1 < N ? gn1 : N - 1;
#pragma unroll
      for (int j = 0; j < 2; ++j) {
        int kp = k0 + kc + 4 * j;
        int kpc = kp < K ? kp : K - 4;
        ull v0 = *(const ull*)(BT + (long)n0c * ldbt + kpc);
        ull v1 = *(const ull*)(BT + (long)n1c * ldbt + kpc);
        rb[0][j] = (gn0 < N && kp < K) ? v0 : 0ull;
        rb[1][j] = (gn1 < N && kp < K) ? v1 : 0ull;
      }
    }
  };

  auto storeLds = [&](int buf) {
#pragma unroll
    for (int j = 0; j < 2; ++j) {
      *(ull*)(&As[buf][rs * LDT + kc + 4 * j]) = ra[0][j];
      *(ull*)(&As[buf][(rs + 64) * LDT + kc + 4 * j]) = ra[1][j];
      *(ull*)(&Bs[buf][rs * LDT + kc + 4 * j]) = rb[0][j];
      *(ull*)(&Bs[buf][(rs + 64) * LDT + kc + 4 * j]) = rb[1][j];
    }
  };

  auto computeTile = [&](int buf) {
    v16bf bfr[2];
#pragma unroll
    for (int nt = 0; nt < 2; ++nt)
      bfr[nt] = frag_b(&Bs[buf][0], wn * 32 + nt * 16, lane);
#pragma unroll
    for (int mt = 0; mt < 4; ++mt) {
      v16bf af = frag_a(&As[buf][0], wm * 64 + mt * 16, lane);
#pragma unroll
      for (int nt = 0; nt < 2; ++nt)
        acc[mt][nt] = __builtin_amdgcn_wmma_f32_16x16x32_bf16(
            false, af, false, bfr[nt], (short)0, acc[mt][nt], false, false);
    }
  };

  loadRegs(0);
  storeLds(0);
  for (int kt = 0; kt < kTiles; ++kt) {
    if (kt + 1 < kTiles) loadRegs(kt + 1);  // next tile's loads in flight
    __syncthreads();                        // stores of buf[kt&1] visible
    computeTile(kt & 1);
    if (kt + 1 < kTiles) storeLds((kt + 1) & 1);
  }

  // ---- epilogue ----
#pragma unroll
  for (int mt = 0; mt < 4; ++mt) {
#pragma unroll
    for (int nt = 0; nt < 2; ++nt) {
      int col = n0 + wn * 32 + nt * 16 + (lane & 15);
#pragma unroll
      for (int r = 0; r < 8; ++r) {
        int row = m0 + wm * 64 + mt * 16 + r + ((lane >> 4) << 3);
        if (row < M && col < N) {
          float v = acc[mt][nt][r] * alpha;
          if (bias) v += bias[col];
          long ci = (long)z * sC + (long)row * ldc + col;
          if constexpr (MODE == 0) {
            ((float*)Cv)[ci] = v;
          } else if constexpr (MODE == 1) {
            ((bf16*)Cv)[ci] = (bf16)v;
          } else if constexpr (MODE == 2) {
            ((bf16*)Cv)[ci] = (bf16)gelu_exact(v);
          } else {
            ((float*)Cv)[ci] = v + res[ci];
          }
        }
      }
    }
  }
}

// ---------------------------------------------------------------------------
// LayerNorm over rows (C channels), fp32 in, bf16 or fp32 out
// ---------------------------------------------------------------------------
template <bool BF16OUT>
__global__ __launch_bounds__(TB)
void ln_rows(const float* __restrict__ x, const float* __restrict__ g,
             const float* __restrict__ b, void* __restrict__ out,
             int C, float eps) {
  const int row = blockIdx.x;
  const float* xr = x + (long)row * C;
  __shared__ float red[TB];
  const int tid = threadIdx.x;

  float s = 0.f;
  for (int c = tid; c < C; c += TB) s += xr[c];
  red[tid] = s; __syncthreads();
  for (int off = TB / 2; off; off >>= 1) {
    if (tid < off) red[tid] += red[tid + off];
    __syncthreads();
  }
  float mean = red[0] / C; __syncthreads();

  float vs = 0.f;
  for (int c = tid; c < C; c += TB) { float d = xr[c] - mean; vs += d * d; }
  red[tid] = vs; __syncthreads();
  for (int off = TB / 2; off; off >>= 1) {
    if (tid < off) red[tid] += red[tid + off];
    __syncthreads();
  }
  float rstd = rsqrtf(red[0] / C + eps); __syncthreads();

  for (int c = tid; c < C; c += TB) {
    float y = (xr[c] - mean) * rstd * g[c] + b[c];
    if (BF16OUT) ((bf16*)out)[(long)row * C + c] = (bf16)y;
    else         ((float*)out)[(long)row * C + c] = y;
  }
}

__global__ void cvt_f32_bf16(const float* __restrict__ in,
                             bf16* __restrict__ out, long n) {
  long i = (long)blockIdx.x * blockDim.x + threadIdx.x;
  long st = (long)gridDim.x * blockDim.x;
  for (; i < n; i += st) out[i] = (bf16)in[i];
}

// Tiled f32 [R,Cc] -> bf16 transpose [Cc,R]; coalesced read and write.
__global__ __launch_bounds__(TB)
void cvt_transpose(const float* __restrict__ in, bf16* __restrict__ out,
                   int R, int Cc) {
  __shared__ float t[32][33];
  const int c0 = blockIdx.x * 32, r0 = blockIdx.y * 32;
  for (int i = threadIdx.y; i < 32; i += 8) {
    int r = r0 + i, c = c0 + threadIdx.x;
    t[i][threadIdx.x] = (r < R && c < Cc) ? in[(long)r * Cc + c] : 0.f;
  }
  __syncthreads();
  for (int i = threadIdx.y; i < 32; i += 8) {
    int c = c0 + i, r = r0 + threadIdx.x;
    if (c < Cc && r < R) out[(long)c * R + r] = (bf16)t[threadIdx.x][i];
  }
}

// Window partition (pad 64x64 -> 5x5 windows of 14x14, zero pad), bf16 -> bf16
__global__ void win_part(const bf16* __restrict__ ln, bf16* __restrict__ out,
                         int C) {
  const int row = blockIdx.x;               // 0..4899
  const int win = row / 196, s = row % 196;
  const int wy = win / 5, wx = win % 5;
  const int sy = s / 14, sx = s % 14;
  const int h = wy * 14 + sy, w = wx * 14 + sx;
  const bool valid = (h < 64) && (w < 64);
  const bf16* src = ln + (long)(h * 64 + w) * C;
  bf16* dst = out + (long)row * C;
  for (int c = threadIdx.x; c < C; c += blockDim.x)
    dst[c] = valid ? src[c] : (bf16)0.0f;
}

// qkv [Mr, 3*nh*hd] -> Q[b][S][hd], K[b][S][hd] (== B^T for QK^T),
//                      Vt[b][hd][S] (== B^T for P@V);   b = win*nh + head
__global__ void split_qkv(const bf16* __restrict__ qkv, bf16* __restrict__ Q,
                          bf16* __restrict__ Kb, bf16* __restrict__ Vt,
                          int S, int nh, int hd) {
  const int row = blockIdx.x;
  const int win = row / S, s = row % S;
  const int C1 = nh * hd, C3 = 3 * C1;
  for (int c = threadIdx.x; c < C1; c += blockDim.x) {
    const int head = c / hd, d = c % hd;
    const long b = (long)win * nh + head;
    Q[(b * S + s) * hd + d]  = qkv[(long)row * C3 + head * hd + d];
    Kb[(b * S + s) * hd + d] = qkv[(long)row * C3 + C1 + head * hd + d];
    Vt[(b * hd + d) * S + s] = qkv[(long)row * C3 + 2 * C1 + head * hd + d];
  }
}

// Decomposed relative bias: out[b][q][kk] = sum_d Q[b][q][d] * tab[pos-kk+dim-1][d]
__global__ void rel_bias(const bf16* __restrict__ Q,
                         const float* __restrict__ tab,
                         float* __restrict__ out, long total, int S, int Wdim,
                         int dim, int isH, int hd) {
  long i = (long)blockIdx.x * blockDim.x + threadIdx.x;
  if (i >= total) return;
  const int kk = (int)(i % dim);
  long t = i / dim;
  const int q = (int)(t % S);
  const long b = t / S;
  const int pos = isH ? (q / Wdim) : (q % Wdim);
  const float* r = tab + (long)(pos - kk + dim - 1) * hd;
  const bf16* qp = Q + (b * S + q) * (long)hd;
  float acc = 0.f;
  for (int d = 0; d < hd; ++d) acc += (float)qp[d] * r[d];
  out[i] = acc;
}

// Row softmax with decomposed rel-bias add, f32 scores -> bf16 probs
__global__ __launch_bounds__(TB)
void softmax_bias(const float* __restrict__ Sc, bf16* __restrict__ P,
                  const float* __restrict__ bH, const float* __restrict__ bW,
                  int S, int Hk, int Wk) {
  const int q = blockIdx.x, b = blockIdx.y;
  const long base = ((long)b * S + q) * S;
  const float* bh = bH + ((long)b * S + q) * Hk;
  const float* bw = bW + ((long)b * S + q) * Wk;
  __shared__ float red[TB];
  const int tid = threadIdx.x;

  float mx = -3.4e38f;
  for (int k = tid; k < S; k += TB)
    mx = fmaxf(mx, Sc[base + k] + bh[k / Wk] + bw[k % Wk]);
  red[tid] = mx; __syncthreads();
  for (int off = TB / 2; off; off >>= 1) {
    if (tid < off) red[tid] = fmaxf(red[tid], red[tid + off]);
    __syncthreads();
  }
  const float m = red[0]; __syncthreads();

  float sum = 0.f;
  for (int k = tid; k < S; k += TB)
    sum += __expf(Sc[base + k] + bh[k / Wk] + bw[k % Wk] - m);
  red[tid] = sum; __syncthreads();
  for (int off = TB / 2; off; off >>= 1) {
    if (tid < off) red[tid] += red[tid + off];
    __syncthreads();
  }
  const float inv = 1.0f / red[0];

  for (int k = tid; k < S; k += TB)
    P[base + k] =
        (bf16)(__expf(Sc[base + k] + bh[k / Wk] + bw[k % Wk] - m) * inv);
}

// Windowed un-partition + head merge: pv[win*nh+head][s][d] -> out[row][head*hd+d]
__global__ void merge_win(const bf16* __restrict__ pv, bf16* __restrict__ out,
                          int nh, int hd) {
  const int row = blockIdx.x;  // h*64+w
  const int h = row >> 6, w = row & 63;
  const int wy = h / 14, sy = h % 14, wx = w / 14, sx = w % 14;
  const int win = wy * 5 + wx, s = sy * 14 + sx;
  for (int c = threadIdx.x; c < nh * hd; c += blockDim.x) {
    const int head = c / hd, d = c % hd;
    out[(long)row * nh * hd + c] =
        pv[(((long)win * nh + head) * 196 + s) * hd + d];
  }
}

// 3x3 SAME im2col, f32 -> bf16, out [H*W][9*C], k ordered (ky,kx,cin) == HWIO
__global__ void im2col3(const float* __restrict__ y, bf16* __restrict__ out,
                        int H, int W, int Cc) {
  const int row = blockIdx.x;
  const int h = row / W, w = row % W;
  for (int c = threadIdx.x; c < 9 * Cc; c += blockDim.x) {
    const int kk = c / Cc, ci = c % Cc;
    const int hh = h + kk / 3 - 1, ww = w + kk % 3 - 1;
    float v = (hh >= 0 && hh < H && ww >= 0 && ww < W)
                  ? y[((long)hh * W + ww) * Cc + ci] : 0.f;
    out[(long)row * 9 * Cc + c] = (bf16)v;
  }
}

// NHWC f32 -> NCHW f32
__global__ void nchw_out(const float* __restrict__ y, float* __restrict__ out,
                         long HW, int Cc) {
  long i = (long)blockIdx.x * blockDim.x + threadIdx.x;
  long st = (long)gridDim.x * blockDim.x;
  long total = HW * Cc;
  for (; i < total; i += st) {
    long p = i / Cc; int c = (int)(i % Cc);
    out[(long)c * HW + p] = y[i];
  }
}

// ---------------------------------------------------------------------------
// Host orchestration
// ---------------------------------------------------------------------------
extern "C" void kernel_launch(void* const* d_in, const int* in_sizes, int n_in,
                              void* d_out, int out_size, void* d_ws,
                              size_t ws_size, hipStream_t stream) {
  (void)in_sizes; (void)n_in; (void)out_size; (void)ws_size;
  const int Mtok = 4096, C = 768, F = 3072, NH = 12, HD = 64;
  const int SW = 196, MRW = 4900;  // windowed: 25 windows of 14x14
  const float scale = 0.125f;      // 64^-0.5

  const float* in_x    = (const float*)d_in[0];
  const float* ln1_g   = (const float*)d_in[1];
  const float* ln1_b   = (const float*)d_in[2];
  const float* qkv_w   = (const float*)d_in[3];
  const float* qkv_b   = (const float*)d_in[4];
  const float* proj_w  = (const float*)d_in[5];
  const float* proj_b  = (const float*)d_in[6];
  const float* rel_h   = (const float*)d_in[7];
  const float* rel_w   = (const float*)d_in[8];
  const float* ln2_g   = (const float*)d_in[9];
  const float* ln2_b   = (const float*)d_in[10];
  const float* fc1_w   = (const float*)d_in[11];
  const float* fc1_b   = (const float*)d_in[12];
  const float* fc2_w   = (const float*)d_in[13];
  const float* fc2_b   = (const float*)d_in[14];
  const float* nk_c1   = (const float*)d_in[15];
  const float* nk_l1g  = (const float*)d_in[16];
  const float* nk_l1b  = (const float*)d_in[17];
  const float* nk_c2   = (const float*)d_in[18];
  const float* nk_l2g  = (const float*)d_in[19];
  const float* nk_l2b  = (const float*)d_in[20];

  // Workspace carve-up (deterministic every call)
  size_t off = 0;
  auto alloc = [&](size_t bytes) {
    char* p = (char*)d_ws + off;
    off = (off + bytes + 255) & ~(size_t)255;
    return p;
  };
  float* xr   = (float*)alloc((size_t)Mtok * C * 4);          // residual stream
  bf16*  lnA  = (bf16*)alloc((size_t)Mtok * C * 2);           // LN out (bf16)
  bf16*  part = (bf16*)alloc((size_t)MRW * C * 2);            // padded windows
  bf16*  qkvb = (bf16*)alloc((size_t)MRW * 3 * C * 2);        // qkv out
  const size_t qkvMax = (size_t)300 * SW * HD;                // >= 12*4096*64
  bf16*  Q    = (bf16*)alloc(qkvMax * 2);
  bf16*  Kb   = (bf16*)alloc(qkvMax * 2);                     // K  [b][S][hd]
  bf16*  Vt   = (bf16*)alloc(qkvMax * 2);                     // V^T [b][hd][S]
  float* sc   = (float*)alloc((size_t)4096 * 4096 * 4);       // scores (or pv)
  bf16*  P    = (bf16*)alloc((size_t)4096 * 4096 * 2);        // probs / mlp / im2col
  float* bHb  = (float*)alloc((size_t)12 * 4096 * 64 * 4);
  float* bWb  = (float*)alloc((size_t)12 * 4096 * 64 * 4);
  bf16*  mrg  = (bf16*)alloc((size_t)Mtok * C * 2);           // merged attn out
  bf16*  wT   = (bf16*)alloc((size_t)C * F * 2);              // bf16 W^T slot
  float* y1   = (float*)alloc((size_t)Mtok * 256 * 4);
  float* y2   = (float*)alloc((size_t)Mtok * 256 * 4);

  auto gemm = [&](int mode, const bf16* A, int lda, long sA, const bf16* BT,
                  int ldbt, long sB, void* Cp, int ldc, long sC,
                  const float* bias, const float* res, int M, int N, int K,
                  int batch, float alpha) {
    dim3 g((N + BN - 1) / BN, (M + BM - 1) / BM, batch), blk(TB);
    switch (mode) {
      case 0: gemm_bf16<0><<<g, blk, 0, stream>>>(A, lda, sA, BT, ldbt, sB, Cp, ldc, sC, bias, res, M, N, K, alpha); break;
      case 1: gemm_bf16<1><<<g, blk, 0, stream>>>(A, lda, sA, BT, ldbt, sB, Cp, ldc, sC, bias, res, M, N, K, alpha); break;
      case 2: gemm_bf16<2><<<g, blk, 0, stream>>>(A, lda, sA, BT, ldbt, sB, Cp, ldc, sC, bias, res, M, N, K, alpha); break;
      default: gemm_bf16<3><<<g, blk, 0, stream>>>(A, lda, sA, BT, ldbt, sB, Cp, ldc, sC, bias, res, M, N, K, alpha); break;
    }
  };
  // weights [K,N] f32 -> [N,K] bf16 (transposed for the GEMM B-operand)
  auto cvtT = [&](const float* src, bf16* dst, int R, int Cc) {
    dim3 g((Cc + 31) / 32, (R + 31) / 32), b(32, 8);
    cvt_transpose<<<g, b, 0, stream>>>(src, dst, R, Cc);
  };

  hipMemcpyAsync(xr, in_x, (size_t)Mtok * C * 4, hipMemcpyDeviceToDevice,
                 stream);

  // ---------------- 4 transformer blocks ----------------
  for (int i = 0; i < 4; ++i) {
    const bool winb = (i < 3);
    const int S = winb ? SW : Mtok;
    const int batch = winb ? 300 : NH;
    const int Mr = winb ? MRW : Mtok;
    const int dim = winb ? 14 : 64;  // H == W of attention input

    // LN1 -> bf16
    ln_rows<true><<<Mtok, TB, 0, stream>>>(xr, ln1_g + i * C, ln1_b + i * C,
                                           lnA, C, 1e-5f);
    const bf16* qkvA = lnA;
    if (winb) {
      win_part<<<MRW, TB, 0, stream>>>(lnA, part, C);
      qkvA = part;
    }

    // qkv = x @ Wqkv + b  (bf16 out);  W^T: [3C][C]
    cvtT(qkv_w + (long)i * C * 3 * C, wT, C, 3 * C);
    gemm(1, qkvA, C, 0, wT, C, 0, qkvb, 3 * C, 0, qkv_b + i * 3 * C, nullptr,
         Mr, 3 * C, C, 1, 1.0f);
    split_qkv<<<Mr, TB, 0, stream>>>(qkvb, Q, Kb, Vt, S, NH, HD);

    // Decomposed relative bias (uses unscaled Q, like reference rq)
    {
      long tot = (long)batch * S * dim;
      int g = (int)((tot + TB - 1) / TB);
      rel_bias<<<g, TB, 0, stream>>>(Q, rel_h + (long)i * 127 * HD, bHb, tot,
                                     S, dim, dim, 1, HD);
      rel_bias<<<g, TB, 0, stream>>>(Q, rel_w + (long)i * 127 * HD, bWb, tot,
                                     S, dim, dim, 0, HD);
    }

    if (winb) {
      // scores = scale * Q @ K^T  (B^T == K natural layout), 300-way batched
      gemm(0, Q, HD, (long)S * HD, Kb, HD, (long)S * HD, sc, S, (long)S * S,
           nullptr, nullptr, S, S, HD, batch, scale);
      softmax_bias<<<dim3(S, batch), TB, 0, stream>>>(sc, P, bHb, bWb, S, dim,
                                                      dim);
      // pv = P @ V  (B^T == V^T), bf16, reuse score buffer
      bf16* pv = (bf16*)sc;
      gemm(1, P, S, (long)S * S, Vt, S, (long)HD * S, pv, HD, (long)S * HD,
           nullptr, nullptr, S, HD, S, batch, 1.0f);
      merge_win<<<Mtok, TB, 0, stream>>>(pv, mrg, NH, HD);
    } else {
      // Global attention, per head (bounds scratch at one 4096x4096 tile);
      // PV epilogue writes straight into the head-merged layout.
      for (int h = 0; h < NH; ++h) {
        const bf16* Qh = Q + (long)h * S * HD;
        const bf16* Kh = Kb + (long)h * S * HD;
        const bf16* Vh = Vt + (long)h * HD * S;
        gemm(0, Qh, HD, 0, Kh, HD, 0, sc, S, 0, nullptr, nullptr, S, S, HD, 1,
             scale);
        softmax_bias<<<dim3(S, 1), TB, 0, stream>>>(
            sc, P, bHb + (long)h * S * dim, bWb + (long)h * S * dim, S, dim,
            dim);
        gemm(1, P, S, 0, Vh, S, 0, mrg + h * HD, C, 0, nullptr, nullptr, S,
             HD, S, 1, 1.0f);
      }
    }

    // x = x + attn @ Wproj + b
    cvtT(proj_w + (long)i * C * C, wT, C, C);
    gemm(3, mrg, C, 0, wT, C, 0, xr, C, 0, proj_b + i * C, xr, Mtok, C, C, 1,
         1.0f);

    // MLP: x = x + fc2(gelu(fc1(ln2(x))))
    ln_rows<true><<<Mtok, TB, 0, stream>>>(xr, ln2_g + i * C, ln2_b + i * C,
                                           lnA, C, 1e-5f);
    bf16* mlpH = P;  // reuse prob buffer
    cvtT(fc1_w + (long)i * C * F, wT, C, F);
    gemm(2, lnA, C, 0, wT, C, 0, mlpH, F, 0, fc1_b + i * F, nullptr, Mtok, F,
         C, 1, 1.0f);
    cvtT(fc2_w + (long)i * F * C, wT, F, C);
    gemm(3, mlpH, F, 0, wT, F, 0, xr, C, 0, fc2_b + i * C, xr, Mtok, C, F, 1,
         1.0f);
  }

  // ---------------- neck ----------------
  cvt_f32_bf16<<<4096, TB, 0, stream>>>(xr, lnA, (long)Mtok * C);
  cvtT(nk_c1, wT, C, 256);
  gemm(0, lnA, C, 0, wT, C, 0, y1, 256, 0, nullptr, nullptr, Mtok, 256, C, 1,
       1.0f);
  ln_rows<false><<<Mtok, TB, 0, stream>>>(y1, nk_l1g, nk_l1b, y2, 256, 1e-6f);

  bf16* col = P;  // reuse
  im2col3<<<Mtok, TB, 0, stream>>>(y2, col, 64, 64, 256);
  cvtT(nk_c2, wT, 9 * 256, 256);
  gemm(0, col, 9 * 256, 0, wT, 9 * 256, 0, y1, 256, 0, nullptr, nullptr, Mtok,
       256, 9 * 256, 1, 1.0f);
  ln_rows<false><<<Mtok, TB, 0, stream>>>(y1, nk_l2g, nk_l2b, y2, 256, 1e-6f);

  nchw_out<<<4096, TB, 0, stream>>>(y2, (float*)d_out, (long)Mtok, 256);
}
